// CausalSelfAttention_21483426414857
// MI455X (gfx1250) — matrix operand order repaired and checked by reading, not compiled, not run on hardware
//
#include <hip/hip_runtime.h>
#include <hip/hip_bf16.h>

typedef _Float16 v16h __attribute__((ext_vector_type(16)));
typedef _Float16 v8h  __attribute__((ext_vector_type(8)));
typedef __fp16   pkh2 __attribute__((ext_vector_type(2)));   // matches cvt_pkrtz return
typedef float    v8f  __attribute__((ext_vector_type(8)));

#define BB   2
#define SS   2048
#define DD   512
#define NH   8
#define NKV  4
#define HDIM 64
#define NTOK (BB*SS)   // 4096
#define LOG2E 1.4426950408889634f

// ---------------------------------------------------------------- WMMA helper
__device__ __forceinline__ v8f wmma_f16(v16h a, v16h b, v8f c) {
  // 8 args: (neg_a, A, neg_b, B, c_mod, C, reuse_a, reuse_b)
  return __builtin_amdgcn_wmma_f32_16x16x32_f16(false, a, false, b, (short)0, c,
                                                false, false);
}

// A-matrix 16x32 f16 fragment from row-major [16 x ld], k-offset kb.
// ISA layout: lane m=L%16; halves 0-7 -> k = (L/16)*8 + 0..7,
//             halves 8-15 -> k = 16 + (L/16)*8 + 0..7.
__device__ __forceinline__ v16h load_a_frag(const _Float16* __restrict__ base,
                                            int ld, int kb, int lane) {
  int m = lane & 15, hs = lane >> 4;
  const _Float16* p = base + (size_t)m * ld + kb + hs * 8;
  v8h lo = *(const v8h*)p;
  v8h hi = *(const v8h*)(p + 16);
  v16h r;
#pragma unroll
  for (int i = 0; i < 8; ++i) { r[i] = lo[i]; r[i + 8] = hi[i]; }
  return r;
}

// B-matrix 32x16 f16 fragment, with B[k][n] stored k-contiguous per n
// (source row n of a row-major [N x ld] matrix, k-offset kb).
// Layout: lane n=L%16; halves 0-7 -> k = (L/16)*8 + 0..7,
//         halves 8-15 -> k = 16 + (L/16)*8 + 0..7.
__device__ __forceinline__ v16h load_b_frag(const _Float16* __restrict__ base,
                                            int ld, int kb, int lane) {
  int n = lane & 15, hs = lane >> 4;
  const _Float16* p = base + (size_t)n * ld + kb + hs * 8;
  v8h lo = *(const v8h*)p;
  v8h hi = *(const v8h*)(p + 16);
  v16h r;
#pragma unroll
  for (int i = 0; i < 8; ++i) { r[i] = lo[i]; r[i + 8] = hi[i]; }
  return r;
}

// ---------------------------------------------------------------- fp32 -> f16
__global__ void __launch_bounds__(256) cvt_kernel(const float* __restrict__ s,
                                                  _Float16* __restrict__ d, int n) {
  int i = blockIdx.x * 256 + threadIdx.x;
  if (i < n) d[i] = (_Float16)s[i];
}

// -------------------------------- V transpose with key-interleave:
// vt[b][kvh][dim][pos], where within each 32-key block, pos 2k   <- key k,
//                                                      pos 2k+1 <- key k+16.
// (P columns use the same permutation, so P.V and row sums are unchanged.)
__global__ void __launch_bounds__(256) transpose_v_kernel(const _Float16* __restrict__ vh,
                                                          _Float16* __restrict__ vt) {
  int i = blockIdx.x * 256 + threadIdx.x;
  if (i >= NTOK * NKV * HDIM) return;
  int pos = i & (SS - 1);
  int r   = i >> 11;               // SS = 2048
  int dim = r & (HDIM - 1);
  int kvh = (r >> 6) & (NKV - 1);
  int b   = r >> 8;
  int wi  = pos & 31, blk = pos >> 5;
  int key = blk * 32 + ((wi & 1) ? (wi >> 1) + 16 : (wi >> 1));
  vt[i] = vh[((size_t)(b * SS + key)) * (NKV * HDIM) + kvh * HDIM + dim];
}

// ------------------------------------------------- GEMM: C[M,N] = A[M,K] * W[N,K]^T
// One wave computes a 32x64 output tile (2 A-frags x 4 B-frags, 8 accumulators).
template <bool HALF_OUT>
__global__ void __launch_bounds__(256) gemm_kernel(const _Float16* __restrict__ A,
                                                   const _Float16* __restrict__ W,
                                                   void* __restrict__ Cv,
                                                   int M, int N, int K) {
  int lane = threadIdx.x & 31;
  int wid  = (blockIdx.x * blockDim.x + threadIdx.x) >> 5;
  int NG = N >> 6;                 // 64-column groups
  int total = (M >> 5) * NG;       // 32-row groups
  if (wid >= total) return;        // uniform per wave: EXEC stays all-ones
  int mt = wid / NG, ng = wid - mt * NG;

  const _Float16* A0 = A + (size_t)(mt * 32) * K;
  const _Float16* A1 = A0 + (size_t)16 * K;
  v8f acc[2][4];
#pragma unroll
  for (int r = 0; r < 2; ++r)
#pragma unroll
    for (int j = 0; j < 4; ++j)
#pragma unroll
      for (int v = 0; v < 8; ++v) acc[r][j][v] = 0.f;

  const _Float16* Wb = W + (size_t)(ng * 64) * K;
  for (int kb = 0; kb < K; kb += 32) {
    v16h a0 = load_a_frag(A0, K, kb, lane);
    v16h a1 = load_a_frag(A1, K, kb, lane);
#pragma unroll
    for (int j = 0; j < 4; ++j) {
      v16h bf = load_b_frag(Wb + (size_t)(j * 16) * K, K, kb, lane);
      acc[0][j] = wmma_f16(a0, bf, acc[0][j]);
      acc[1][j] = wmma_f16(a1, bf, acc[1][j]);
    }
  }

  int n0 = ng * 64 + (lane & 15);
  int hs = lane >> 4;
#pragma unroll
  for (int r = 0; r < 2; ++r)
#pragma unroll
    for (int v = 0; v < 8; ++v) {
      size_t row = (size_t)(mt * 32 + r * 16 + hs * 8 + v) * N;
      if constexpr (HALF_OUT) {
        _Float16* C = (_Float16*)Cv;
#pragma unroll
        for (int j = 0; j < 4; ++j) C[row + n0 + j * 16] = (_Float16)acc[r][j][v];
      } else {
        float* C = (float*)Cv;
#pragma unroll
        for (int j = 0; j < 4; ++j) C[row + n0 + j * 16] = acc[r][j][v];
      }
    }
}

// ------------------------------------------------- RoPE + RMS-norm (in place, f16)
// One wave per (token, head); lane j handles dim pair (j, j+32).
// extra: additional scale folded into the output (1/sqrt(HD) for q).
__global__ void __launch_bounds__(256) rope_rms_kernel(_Float16* __restrict__ buf,
                                                       int tokens, int nheads,
                                                       const float* __restrict__ gainp,
                                                       int usegain, float extra) {
  int lane = threadIdx.x & 31;
  int wid  = (blockIdx.x * blockDim.x + threadIdx.x) >> 5;
  int total = tokens * nheads;
  if (wid >= total) return;
  int head = wid % nheads, row = wid / nheads;
  int s = row & (SS - 1);          // position within sequence
  int ld = nheads * HDIM;
  _Float16* p = buf + (size_t)row * ld + head * HDIM;

  float x1 = (float)p[lane];
  float x2 = (float)p[lane + 32];
  float inv = __powf(10000.0f, -(float)lane / 32.0f);
  float fr  = (float)s * inv;
  float c = __cosf(fr), sn = __sinf(fr);
  float r1 =  x1 * c + x2 * sn;
  float r2 = -x1 * sn + x2 * c;

  float ss = r1 * r1 + r2 * r2;
#pragma unroll
  for (int m = 1; m < 32; m <<= 1) ss += __shfl_xor(ss, m, 32);
  float g = (usegain ? gainp[0] : 1.0f) * extra;
  float rn = rsqrtf(ss * (1.0f / 64.0f) + 1e-6f) * g;

  p[lane]      = (_Float16)(r1 * rn);
  p[lane + 32] = (_Float16)(r2 * rn);
}

// ------------------------------------------------- Flash attention block step
// One 32-key block: scores via WMMA, exp with STATIC shift (no running max:
// |score| <= 8*gain by Cauchy-Schwarz on the RMS-normalized q/k), row sums via
// a ones-matrix WMMA, P.V via WMMA. MASK only for the final (diagonal) block.
// nsh = -shift*log2(e); p = exp2(fma(a, log2e, nsh)) -> v_fma + v_exp only.
template <bool MASK>
__device__ __forceinline__ void attn_block(int kb, int qbase, int lane, int lane16, int hs,
                                           const v16h& qa0, const v16h& qa1, const v16h& ones,
                                           const _Float16* __restrict__ khb,
                                           const _Float16* __restrict__ vtb,
                                           _Float16* __restrict__ pl, float nsh,
                                           v8f& rs, v8f& o0, v8f& o1, v8f& o2, v8f& o3) {
  // ---- scores S = Q(16x64) . K_block(32x64)^T -> two 16x16 frags
  v8f z;
#pragma unroll
  for (int v = 0; v < 8; ++v) z[v] = 0.f;
  const _Float16* Kb = khb + (size_t)kb * (NKV * HDIM);
  v8f s0 = wmma_f16(qa0, load_b_frag(Kb,                      NKV * HDIM, 0, lane), z);
  v8f s1 = wmma_f16(qa0, load_b_frag(Kb + 16 * (NKV * HDIM),  NKV * HDIM, 0, lane), z);
  s0 = wmma_f16(qa1, load_b_frag(Kb + 32,                     NKV * HDIM, 0, lane), s0);
  s1 = wmma_f16(qa1, load_b_frag(Kb + 32 + 16 * (NKV * HDIM), NKV * HDIM, 0, lane), s1);

  // ---- P = exp(S - shift), stored as packed f16 pairs (interleaved key layout)
#pragma unroll
  for (int v = 0; v < 8; ++v) {
    float a0 = s0[v];
    float a1 = s1[v];
    if (MASK) {
      int qi = qbase + hs * 8 + v;
      if (kb + lane16 > qi)      a0 = -3.0e38f;   // causal mask -> exp -> 0
      if (kb + 16 + lane16 > qi) a1 = -3.0e38f;
    }
    float p0 = __builtin_amdgcn_exp2f(fmaf(a0, LOG2E, nsh));
    float p1 = __builtin_amdgcn_exp2f(fmaf(a1, LOG2E, nsh));
    int m = hs * 8 + v;
    // pos 2*lane16 <-> key lane16 ; pos 2*lane16+1 <-> key lane16+16 (matches vt)
    *(pkh2*)(pl + m * 32 + 2 * lane16) = __builtin_amdgcn_cvt_pkrtz(p0, p1);
  }
  asm volatile("s_wait_dscnt 0" ::: "memory");

  // ---- reload P as an A-matrix fragment (16x32)
  const _Float16* pp = pl + lane16 * 32;
  v8h l0 = *(const v8h*)(pp + hs * 8);
  v8h l1 = *(const v8h*)(pp + 16 + hs * 8);
  v16h pa;
#pragma unroll
  for (int i = 0; i < 8; ++i) { pa[i] = l0[i]; pa[i + 8] = l1[i]; }
  asm volatile("" ::: "memory");

  // ---- row sums: P(16x32) . ones(32x16) -> same layout as O accumulators
  rs = wmma_f16(pa, ones, rs);

  // ---- O += P(16x32) . V_block(32x64): 4 dim-tiles, contiguous loads from vT
  o0 = wmma_f16(pa, load_b_frag(vtb + (size_t)0  * SS, SS, kb, lane), o0);
  o1 = wmma_f16(pa, load_b_frag(vtb + (size_t)16 * SS, SS, kb, lane), o1);
  o2 = wmma_f16(pa, load_b_frag(vtb + (size_t)32 * SS, SS, kb, lane), o2);
  o3 = wmma_f16(pa, load_b_frag(vtb + (size_t)48 * SS, SS, kb, lane), o3);
}

// ------------------------------------------------- Flash attention (causal, GQA)
// One wave per (b, h, 16-query tile). Key blocks of 32.
__global__ void __launch_bounds__(256) attn_kernel(const _Float16* __restrict__ qh,
                                                   const _Float16* __restrict__ kh,
                                                   const _Float16* __restrict__ vt,
                                                   _Float16* __restrict__ oh,
                                                   const float* __restrict__ gainp) {
  __shared__ __align__(64) _Float16 plds[8][16 * 32];  // P scratch, 1KB/wave
  int lane = threadIdx.x & 31;
  int w    = threadIdx.x >> 5;
  int wid  = blockIdx.x * 8 + w;
  int qt = wid & 127;              // S/16 = 128 query tiles
  int h  = (wid >> 7) & 7;
  int b  = wid >> 10;
  int qbase = qt << 4;
  int kvh = h >> 1;                // GQA: rep = H/KV = 2
  int lane16 = lane & 15, hs = lane >> 4;

  // |score| <= ||q||*||k|| = (8*gain*0.125)*8 = 8*gain.  Shift keeps p <= e^11
  // (f16-safe) while leaving typical p values near 1 (softmax shift-invariant).
  float bound = 8.0f * gainp[0];
  float shift = fmaxf(bound - 11.0f, 0.0f);
  float nsh   = -shift * LOG2E;

  const _Float16* Qb = qh + ((size_t)(b * SS + qbase)) * (NH * HDIM) + h * HDIM;
  v16h qa0 = load_a_frag(Qb, NH * HDIM, 0,  lane);   // dims 0..31  (pre-scaled by gain/8)
  v16h qa1 = load_a_frag(Qb, NH * HDIM, 32, lane);   // dims 32..63

  const _Float16* khb = kh + ((size_t)(b * SS)) * (NKV * HDIM) + kvh * HDIM;
  const _Float16* vtb = vt + ((size_t)(b * NKV + kvh)) * HDIM * SS;
  _Float16* pl = &plds[w][0];

  v16h ones;
#pragma unroll
  for (int i = 0; i < 16; ++i) ones[i] = (_Float16)1.0f;

  v8f rs, o0, o1, o2, o3;
#pragma unroll
  for (int v = 0; v < 8; ++v) {
    rs[v] = 0.f; o0[v] = 0.f; o1[v] = 0.f; o2[v] = 0.f; o3[v] = 0.f;
  }

  int nb = ((qbase + 15) >> 5) + 1;
  for (int blk = 0; blk < nb - 1; ++blk)          // mask-free body (kb+31 <= qbase-1)
    attn_block<false>(blk << 5, qbase, lane, lane16, hs, qa0, qa1, ones,
                      khb, vtb, pl, nsh, rs, o0, o1, o2, o3);
  attn_block<true>((nb - 1) << 5, qbase, lane, lane16, hs, qa0, qa1, ones,
                   khb, vtb, pl, nsh, rs, o0, o1, o2, o3);

  // ---- normalize and store (f16) into oh[(b,s), h*64 + d]
#pragma unroll
  for (int v = 0; v < 8; ++v) {
    float inv = __builtin_amdgcn_rcpf(rs[v]);
    int tok = qbase + hs * 8 + v;
    _Float16* op = oh + ((size_t)(b * SS + tok)) * (NH * HDIM) + h * HDIM + lane16;
    op[0]  = (_Float16)(o0[v] * inv);
    op[16] = (_Float16)(o1[v] * inv);
    op[32] = (_Float16)(o2[v] * inv);
    op[48] = (_Float16)(o3[v] * inv);
  }
}

// ---------------------------------------------------------------- launcher
extern "C" void kernel_launch(void* const* d_in, const int* in_sizes, int n_in,
                              void* d_out, int out_size, void* d_ws, size_t ws_size,
                              hipStream_t stream) {
  (void)in_sizes; (void)n_in; (void)out_size; (void)ws_size;
  const float* x    = (const float*)d_in[0];
  const float* wq   = (const float*)d_in[1];
  const float* wk   = (const float*)d_in[2];
  const float* wv   = (const float*)d_in[3];
  const float* wo   = (const float*)d_in[4];
  const float* gain = (const float*)d_in[5];

  _Float16* xh  = (_Float16*)d_ws;
  _Float16* wqh = xh  + (size_t)NTOK * DD;
  _Float16* wkh = wqh + (size_t)DD * DD;
  _Float16* wvh = wkh + (size_t)(NKV * HDIM) * DD;
  _Float16* woh = wvh + (size_t)(NKV * HDIM) * DD;
  _Float16* qh  = woh + (size_t)DD * DD;
  _Float16* kh  = qh  + (size_t)NTOK * DD;
  _Float16* vh  = kh  + (size_t)NTOK * (NKV * HDIM);
  _Float16* oh  = vh  + (size_t)NTOK * (NKV * HDIM);
  _Float16* vt  = oh  + (size_t)NTOK * DD;

  cvt_kernel<<<(NTOK * DD + 255) / 256, 256, 0, stream>>>(x, xh, NTOK * DD);
  cvt_kernel<<<(DD * DD + 255) / 256, 256, 0, stream>>>(wq, wqh, DD * DD);
  cvt_kernel<<<(NKV * HDIM * DD + 255) / 256, 256, 0, stream>>>(wk, wkh, NKV * HDIM * DD);
  cvt_kernel<<<(NKV * HDIM * DD + 255) / 256, 256, 0, stream>>>(wv, wvh, NKV * HDIM * DD);
  cvt_kernel<<<(DD * DD + 255) / 256, 256, 0, stream>>>(wo, woh, DD * DD);

  {
    int tiles = (NTOK / 32) * (DD / 64);     // 1024 waves
    gemm_kernel<true><<<(tiles * 32 + 255) / 256, 256, 0, stream>>>(xh, wqh, qh, NTOK, DD, DD);
  }
  {
    int tiles = (NTOK / 32) * ((NKV * HDIM) / 64);   // 512 waves
    gemm_kernel<true><<<(tiles * 32 + 255) / 256, 256, 0, stream>>>(xh, wkh, kh, NTOK, NKV * HDIM, DD);
    gemm_kernel<true><<<(tiles * 32 + 255) / 256, 256, 0, stream>>>(xh, wvh, vh, NTOK, NKV * HDIM, DD);
  }

  // q gets gain * 1/sqrt(HD) folded in; k gets plain RMS norm.
  rope_rms_kernel<<<(NTOK * NH * 32 + 255) / 256, 256, 0, stream>>>(qh, NTOK, NH, gain, 1, 0.125f);
  rope_rms_kernel<<<(NTOK * NKV * 32 + 255) / 256, 256, 0, stream>>>(kh, NTOK, NKV, gain, 0, 1.0f);

  transpose_v_kernel<<<(NTOK * NKV * HDIM + 255) / 256, 256, 0, stream>>>(vh, vt);

  attn_kernel<<<(BB * NH * (SS / 16)) / 8, 256, 0, stream>>>(qh, kh, vt, oh, gain);

  {
    int tiles = (NTOK / 32) * (DD / 64);
    gemm_kernel<false><<<(tiles * 32 + 255) / 256, 256, 0, stream>>>(oh, woh, (float*)d_out, NTOK, DD, DD);
  }
}
